// BSplineLayer_40054865002951
// MI455X (gfx1250) — matrix operand order repaired and checked by reading, not compiled, not run on hardware
//
#include <hip/hip_runtime.h>

// ---------------------------------------------------------------------------
// BSpline pipeline for MI455X (gfx1250, wave32):
//   prep:   Ab = bf16(tanh(W1+b1)) [4096x2048], Bb = bf16(W2) [2048x2048]
//   gemm:   c = Ab @ Bb via v_wmma_f32_16x16x32_bf16; A tile staged by the
//           Tensor Data Mover (tensor_load_to_lds + s_wait_tensorcnt), fused
//           epilogue gpart[m][nblk] = sum_n tanh(c+b2[n]) * W3[n]
//   reduce: u[m] = 3 * tanh(sum_nblk gpart + b3)           (64x64 table)
//   spline: per-point quadratic B-spline; u broadcast to LDS with
//           global_load_async_to_lds_b128 + s_wait_asynccnt
// ---------------------------------------------------------------------------

typedef __attribute__((ext_vector_type(16))) __bf16 v16bf;
typedef __attribute__((ext_vector_type(8)))  float  v8f;
typedef __attribute__((ext_vector_type(4)))  int    v4i;
typedef __attribute__((ext_vector_type(8)))  int    v8i;

#define MDIM 4096
#define NDIM 2048
#define KDIM 2048
#define NBLK 16            // NDIM / TILE_N
#define TILE_M 128
#define TILE_N 128
#define TILE_K 64
#define LPITCH 72          // LDS row pitch in bf16 units (64 data + 8 pad)
#define NPOINTS 2097152

__device__ __forceinline__ unsigned short f32_to_bf16(float f) {
  unsigned int u = __float_as_uint(f);
  u += 0x7FFFu + ((u >> 16) & 1u);        // round-to-nearest-even
  return (unsigned short)(u >> 16);
}

// ---------------- prep: h1 = tanh(W1 + b1) -> bf16 ; W2 -> bf16 -------------
__global__ __launch_bounds__(256) void prep_kernel(
    const float* __restrict__ W1, const float* __restrict__ b1,
    const float* __restrict__ W2,
    unsigned short* __restrict__ Ab, unsigned short* __restrict__ Bb) {
  unsigned int i = blockIdx.x * 256u + threadIdx.x;   // exactly MDIM*KDIM threads
  float v = tanhf(W1[i] + b1[i & (NDIM - 1u)]);
  Ab[i] = f32_to_bf16(v);
  if (i < (unsigned)(KDIM * NDIM)) Bb[i] = f32_to_bf16(W2[i]);
}

// ---------------- bf16 WMMA GEMM with TDM A-staging -------------------------
__global__ __launch_bounds__(256) void gemm_kernel(
    const unsigned short* __restrict__ Ab,   // MDIM x KDIM  (bf16 bits)
    const unsigned short* __restrict__ Bb,   // KDIM x NDIM  (bf16 bits)
    const float* __restrict__ b2,            // NDIM
    const float* __restrict__ W3,            // NDIM
    float* __restrict__ gpart) {             // MDIM x NBLK partial row sums
  __shared__ unsigned short As[TILE_M * LPITCH];   // row-major  As[m][k] (TDM-padded)
  __shared__ unsigned short Bs[TILE_N * LPITCH];   // transposed Bs[n][k]
  __shared__ float red[TILE_M * 33];

  const unsigned tid  = threadIdx.x;
  const unsigned lane = tid & 31u;
  const unsigned wave = tid >> 5;
  const unsigned g    = lane >> 4;     // half-wave (K-group select)
  const unsigned ln   = lane & 15u;    // row/col within 16
  const unsigned wrow = wave >> 1;     // 0..3 -> 32 M-rows each
  const unsigned wcol = wave & 1u;     // 0..1 -> 64 N-cols each
  const unsigned mbase = blockIdx.y * TILE_M;
  const unsigned nbase = blockIdx.x * TILE_N;

  // wave-relative LDS byte address of As (generic pointer low 32 bits)
  const unsigned ldsAs = (unsigned)(unsigned long long)(const void*)As;

  const v8f vzero = {0.f,0.f,0.f,0.f,0.f,0.f,0.f,0.f};
  v8f acc[2][4];
#pragma unroll
  for (int mf = 0; mf < 2; ++mf)
#pragma unroll
    for (int nf = 0; nf < 4; ++nf) acc[mf][nf] = vzero;

  for (unsigned k0 = 0; k0 < KDIM; k0 += TILE_K) {
    // ---- stage A tile (128 x 64 bf16) with the Tensor Data Mover ----------
    // D# group0: count=1 | lds_addr | global_addr | type=2
    // D# group1: data_size=2B, pad every 32 dwords by 4 dwords -> pitch 72,
    //            tensor_dim0=K, tensor_dim1=M, tile 64x128, stride K
    if (wave == 0) {
      unsigned long long ga =
          (unsigned long long)(const void*)(Ab + (size_t)mbase * KDIM + k0);
      v4i g0;
      g0[0] = 1;                                          // count=1, user D#
      g0[1] = (int)ldsAs;                                 // lds_addr (bytes)
      g0[2] = (int)(unsigned)(ga & 0xFFFFFFFFull);        // global_addr[31:0]
      g0[3] = (int)((unsigned)((ga >> 32) & 0x1FFFFFFu) | (2u << 30)); // type=2
      v8i g1;
      g1[0] = (1 << 16)          // data_size = 2 bytes
            | (1 << 20)          // pad_enable
            | (4 << 22)          // pad_interval: 32 dwords (64 bf16)
            | (3 << 25);         // pad_amount:   4 dwords  (8 bf16)
      g1[1] = (KDIM & 0xFFFF) << 16;                      // tensor_dim0 lo
      g1[2] = (KDIM >> 16) | ((MDIM & 0xFFFF) << 16);     // dim0 hi | dim1 lo
      g1[3] = (MDIM >> 16) | (TILE_K << 16);              // dim1 hi | tile_dim0
      g1[4] = TILE_M;                                     // tile_dim1 (tile_dim2=0)
      g1[5] = KDIM;                                       // tensor_dim0_stride lo
      g1[6] = 0;
      g1[7] = 0;
      asm volatile("tensor_load_to_lds %0, %1"
                   :: "s"(g0), "s"(g1) : "memory");
      __builtin_amdgcn_s_wait_tensorcnt(0);
    }
    // ---- stage B tile transposed: read row-major 64 x 128, write Bs[n][k] --
    {
      unsigned kk   = tid >> 2;            // 0..63
      unsigned nsub = (tid & 3u) * 32u;    // 0,32,64,96
      const unsigned short* src = Bb + (size_t)(k0 + kk) * NDIM + nbase + nsub;
      alignas(16) unsigned short tmp[32];
#pragma unroll
      for (int j = 0; j < 4; ++j)
        *(uint4*)(tmp + j * 8) = *(const uint4*)(src + j * 8);
#pragma unroll
      for (int i = 0; i < 32; ++i)
        Bs[(nsub + i) * LPITCH + kk] = tmp[i];
    }
    __syncthreads();

    // prefetch next B K-tile while this one computes (global_prefetch_b8)
    if (k0 + TILE_K < KDIM) {
      __builtin_prefetch(Bb + (size_t)(k0 + TILE_K + (tid >> 2)) * NDIM +
                             nbase + (tid & 3u) * 32u, 0, 0);
    }

    // two WMMA K-fragments of depth 32
#pragma unroll
    for (int ks = 0; ks < TILE_K; ks += 32) {
      union Frag { uint4 q[2]; v16bf v; };
      Frag a[2], b[4];
#pragma unroll
      for (int mf = 0; mf < 2; ++mf) {
        // 16-bit A 16x32 layout: lane half g -> K groups {g*8.., 16+g*8..}
        const unsigned short* ap =
            As + (wrow * 32u + mf * 16u + ln) * LPITCH + ks + g * 8u;
        a[mf].q[0] = *(const uint4*)(ap);
        a[mf].q[1] = *(const uint4*)(ap + 16);
      }
#pragma unroll
      for (int nf = 0; nf < 4; ++nf) {
        // 16-bit B 32x16 layout: lane half g -> K groups {g*8.., 16+g*8..}
        const unsigned short* bp =
            Bs + (wcol * 64u + nf * 16u + ln) * LPITCH + ks + g * 8u;
        b[nf].q[0] = *(const uint4*)(bp);
        b[nf].q[1] = *(const uint4*)(bp + 16);
      }
#pragma unroll
      for (int mf = 0; mf < 2; ++mf)
#pragma unroll
        for (int nf = 0; nf < 4; ++nf)
          acc[mf][nf] = __builtin_amdgcn_wmma_f32_16x16x32_bf16(
              false, a[mf].v, false, b[nf].v, (short)0, acc[mf][nf],
              false, false);
    }
    __syncthreads();
  }

  // fused epilogue: tanh(c + b2[n]) * W3[n], deterministic LDS tree reduce
#pragma unroll
  for (int mf = 0; mf < 2; ++mf) {
    float s[8];
#pragma unroll
    for (int r = 0; r < 8; ++r) s[r] = 0.f;
#pragma unroll
    for (int nf = 0; nf < 4; ++nf) {
      unsigned nG = nbase + wcol * 64u + nf * 16u + ln;
      float bb = b2[nG];
      float ww = W3[nG];
#pragma unroll
      for (int r = 0; r < 8; ++r)
        s[r] += tanhf(acc[mf][nf][r] + bb) * ww;
    }
    unsigned slot = wcol * 16u + ln;             // 32 N-slots per M-row
#pragma unroll
    for (int r = 0; r < 8; ++r) {
      unsigned m = wrow * 32u + mf * 16u + g * 8u + r;  // C/D layout: M = r + 8*g
      red[m * 33u + slot] = s[r];
    }
  }
  __syncthreads();
  if (tid < TILE_M) {
    float s = 0.f;
#pragma unroll
    for (int j = 0; j < 32; ++j) s += red[tid * 33u + j];
    gpart[(size_t)(mbase + tid) * NBLK + blockIdx.x] = s;
  }
}

// ---------------- reduce: u[m] = 3*tanh(rowsum + b3) ------------------------
__global__ __launch_bounds__(256) void reduce_kernel(
    const float* __restrict__ gpart, const float* __restrict__ b3,
    float* __restrict__ u) {
  unsigned m = blockIdx.x * 256u + threadIdx.x;    // 0..4095
  float s = 0.f;
#pragma unroll
  for (int j = 0; j < NBLK; ++j) s += gpart[m * NBLK + j];
  u[m] = 3.0f * tanhf(s + b3[0]);
}

// ---------------- spline: per-point gather from LDS-resident u --------------
__global__ __launch_bounds__(256) void spline_kernel(
    const float* __restrict__ pts, const float* __restrict__ u,
    float* __restrict__ out) {
  __shared__ float us[64 * 64];

  // broadcast 16KB table into LDS via async copy (no VGPR round-trip)
  {
    unsigned lbase = (unsigned)(unsigned long long)(const void*)us;
    unsigned long long gbase = (unsigned long long)(const void*)u;
#pragma unroll
    for (int j = 0; j < 4; ++j) {
      unsigned off = (threadIdx.x + (unsigned)j * 256u) * 16u;
      unsigned ldsa = lbase + off;
      unsigned long long ga = gbase + off;
      asm volatile("global_load_async_to_lds_b128 %0, %1, off"
                   :: "v"(ldsa), "v"(ga) : "memory");
    }
    asm volatile("s_wait_asynccnt 0x0" ::: "memory");
  }
  __syncthreads();

  unsigned idx = blockIdx.x * 256u + threadIdx.x;
  float2 pt = ((const float2*)pts)[idx];
  float x = (pt.x + 1.0f) * 0.5f;
  float y = pt.y;

  float px = x * 62.0f;
  float fx = floorf(px);
  int posx = (int)fx + 1; if (posx > 62) posx = 62;
  float tx = px - fx;
  float ox = 1.0f - tx;
  float bx0 = 0.5f * ox * ox;
  float bx1 = tx - tx * tx + 0.5f;
  float bx2 = 0.5f * tx * tx;

  float py = y * 62.0f;
  float fy = floorf(py);
  int posy = (int)fy + 1; if (posy > 62) posy = 62;
  float ty = py - fy;
  float oy = 1.0f - ty;
  float by0 = 0.5f * oy * oy;
  float by1 = ty - ty * ty + 0.5f;
  float by2 = 0.5f * ty * ty;

  const float* row0 = us + (posx - 1) * 64 + (posy - 1);
  const float* row1 = row0 + 64;
  const float* row2 = row1 + 64;
  float r0 = fmaf(row0[0], by0, fmaf(row0[1], by1, row0[2] * by2));
  float r1 = fmaf(row1[0], by0, fmaf(row1[1], by1, row1[2] * by2));
  float r2 = fmaf(row2[0], by0, fmaf(row2[1], by1, row2[2] * by2));
  out[idx] = fmaf(r0, bx0, fmaf(r1, bx1, r2 * bx2));
}

// ---------------------------------------------------------------------------
extern "C" void kernel_launch(void* const* d_in, const int* in_sizes, int n_in,
                              void* d_out, int out_size, void* d_ws, size_t ws_size,
                              hipStream_t stream) {
  (void)in_sizes; (void)n_in; (void)out_size; (void)ws_size;
  const float* points = (const float*)d_in[0];
  const float* W1 = (const float*)d_in[1];
  const float* b1 = (const float*)d_in[2];
  const float* W2 = (const float*)d_in[3];
  const float* b2 = (const float*)d_in[4];
  const float* W3 = (const float*)d_in[5];
  const float* b3 = (const float*)d_in[6];
  float* out = (float*)d_out;

  // workspace layout: Ab 16MB | Bb 8MB | gpart 256KB | u 16KB  (~24.3MB total)
  char* ws = (char*)d_ws;
  unsigned short* Ab = (unsigned short*)ws;
  unsigned short* Bb = (unsigned short*)(ws + ((size_t)16 << 20));
  float* gpart = (float*)(ws + ((size_t)24 << 20));
  float* u     = (float*)(ws + ((size_t)24 << 20) + (size_t)MDIM * NBLK * 4);

  prep_kernel<<<dim3((MDIM * KDIM) / 256), 256, 0, stream>>>(W1, b1, W2, Ab, Bb);
  gemm_kernel<<<dim3(NDIM / TILE_N, MDIM / TILE_M), 256, 0, stream>>>(
      Ab, Bb, b2, W3, gpart);
  reduce_kernel<<<dim3(MDIM / 256), 256, 0, stream>>>(gpart, b3, u);
  spline_kernel<<<dim3(NPOINTS / 256), 256, 0, stream>>>(points, u, out);
}
